// GWM_7876970021391
// MI455X (gfx1250) — matrix operand order repaired and checked by reading, not compiled, not run on hardware
//
#include <hip/hip_runtime.h>
#include <math.h>

// Problem constants (match reference)
#define Nv  50000
#define Gg  1000
#define Ee  640000
#define Hh  128
#define Kh  8
#define BFe 11
#define H3  (3*Hh)
#define MT  4              // row-tiles (of 16) per block: 64-row M-tile

typedef __attribute__((ext_vector_type(16))) __bf16 v16bf;
typedef __attribute__((ext_vector_type(8)))  float  v8f;

__device__ __forceinline__ float sigmf(float x) { return 1.f / (1.f + __expf(-x)); }

__device__ __forceinline__ float activate(float x, int act) {
    if (act == 1) return tanhf(x);
    if (act == 2) return sigmf(x);
    if (act == 3) return x > 0.f ? x : 0.1f * x;
    return x;
}

__device__ __forceinline__ unsigned pack_bf2(float a, float b) {
    unsigned short u0 = __builtin_bit_cast(unsigned short, (__bf16)a);
    unsigned short u1 = __builtin_bit_cast(unsigned short, (__bf16)b);
    return (unsigned)u0 | ((unsigned)u1 << 16);
}

// Guarded A element fetch: optional row-gather and [A0 | A1] concat along K.
__device__ __forceinline__ float ldA(const float* __restrict__ A0,
                                     const int* __restrict__ gatherA, int A0cols,
                                     const float* __restrict__ A1, int A1cols,
                                     int Ktot, int row, int gk, int M) {
    if (row >= M) return 0.f;
    if (gk < A0cols) {
        int ar = gatherA ? gatherA[row] : row;
        return A0[(size_t)ar * A0cols + gk];
    }
    if (gk < Ktot) return A1[(size_t)row * A1cols + (gk - A0cols)];
    return 0.f;
}

// ---------------------------------------------------------------------------
// Core: one block (256 threads = 8 wave32) computes a 64-row x 128-col output
// tile of (A @ B). Each wave owns a 16-col slice and MT=4 row-tiles, reusing
// one B fragment across 4 WMMAs per k-step (4x compute density, 4x less B
// re-staging). Tiles are staged to LDS as bf16 in WMMA fragment order
// (frag[lane][0..15]): each lane reads a whole operand fragment as two
// ds_load_b128's. Double-buffered LDS: one barrier per k-tile; next tile's
// global loads / LDS stores overlap the current WMMAs.
// FAST=true: no gather, no concat, Ktot % 32 == 0 -> branchless staging
// (row index clamped; out-of-tile rows are never stored).
// ---------------------------------------------------------------------------
template <bool FAST>
__device__ __forceinline__ void wmma_rowtile(
    const float* __restrict__ A0, const int* __restrict__ gatherA, int A0cols,
    const float* __restrict__ A1, int A1cols,
    const float* __restrict__ B, int ldb, int cb,
    int rowbase, int M, v8f acc[MT])
{
    __shared__ v16bf Afrag[2][MT][32];   // per row-tile, per-lane A fragments (8 KB)
    __shared__ v16bf Bfrag[2][8][32];    // per-wave, per-lane B fragments (16 KB)

    const int tid  = threadIdx.x;
    const int wave = tid >> 5;
    const int lane = tid & 31;
    const int Ktot = FAST ? A0cols : (A0cols + A1cols);
    const int kTiles = (Ktot + 31) >> 5;

    // Stage k-tile kt into LDS buffer buf, pre-permuted to fragment order.
    auto stage = [&](int kt, int buf) {
        const int k0 = kt << 5;
        // ---- A: 64 rows x 32 k; bf16 pairs, coalesced along k ----
#pragma unroll
        for (int t = 0; t < MT; ++t) {
            int idx = tid + t * 256;
            int r   = idx >> 4;          // row-in-tile 0..63
            int k   = (idx & 15) << 1;   // even k 0..30
            int row = rowbase + r;
            float v0, v1;
            if (FAST) {
                int rr = row < M ? row : M - 1;          // branchless clamp
                const float* p = A0 + (size_t)rr * A0cols + (k0 + k);
                v0 = p[0];
                v1 = p[1];
            } else {
                v0 = ldA(A0, gatherA, A0cols, A1, A1cols, Ktot, row, k0 + k, M);
                v1 = ldA(A0, gatherA, A0cols, A1, A1cols, Ktot, row, k0 + k + 1, M);
            }
            // consumer (row-tile mt, lane L, slot j):
            int mt    = r >> 4;
            int rr16  = r & 15;
            int hi    = k >> 4;                   // upper 16 of K
            int base  = k & 15;
            int halfL = base >> 3;                // lane half
            int j     = (base & 7) + (hi << 3);   // even
            int L     = (halfL << 4) + rr16;
            *(unsigned*)((__bf16*)&Afrag[buf][mt][L] + j) = pack_bf2(v0, v1);
        }
        // ---- B: 32 k x 128 c; bf16 k-pairs, coalesced over c ----
#pragma unroll
        for (int i = tid; i < 16 * 128; i += 256) {
            int kk = (i >> 7) << 1;      // even k-in-tile 0..30
            int c  = i & 127;
            int g0 = k0 + kk;
            float v0, v1;
            if (FAST) {
                const float* p = B + (size_t)g0 * ldb + cb + c;
                v0 = p[0];
                v1 = p[ldb];
            } else {
                v0 = (g0     < Ktot) ? B[(size_t)g0       * ldb + cb + c] : 0.f;
                v1 = (g0 + 1 < Ktot) ? B[(size_t)(g0 + 1) * ldb + cb + c] : 0.f;
            }
            int j     = (kk & 7) + ((kk >> 4) << 3);   // even
            int halfL = (kk >> 3) & 1;
            int wv    = c >> 4;
            int L     = (halfL << 4) + (c & 15);
            *(unsigned*)((__bf16*)&Bfrag[buf][wv][L] + j) = pack_bf2(v0, v1);
        }
    };

#pragma unroll
    for (int mt = 0; mt < MT; ++mt) acc[mt] = (v8f){};
    stage(0, 0);
    for (int kt = 0; kt < kTiles; ++kt) {
        __syncthreads();                             // buffer kt&1 ready
        v16bf bv = Bfrag[kt & 1][wave][lane];        // 2x ds_load_b128, reused 4x
        if (kt + 1 < kTiles) {
            if (((kt + 2) << 5) < Ktot)              // prefetch tile kt+2 of B
                __builtin_prefetch(B + (size_t)((kt + 2) << 5) * ldb + cb + (tid & 127), 0, 1);
            stage(kt + 1, (kt + 1) & 1);             // overlaps the wmmas below
        }
#pragma unroll
        for (int mt = 0; mt < MT; ++mt) {
            v16bf af = Afrag[kt & 1][mt][lane];      // 2x ds_load_b128
            acc[mt] = __builtin_amdgcn_wmma_f32_16x16x32_bf16(
                false, af, false, bv, (short)0, acc[mt], false, false);
        }
    }
}

// ---------------------------------------------------------------------------
// Generic fused GEMM: C = act(gather(A0)|A1 @ B + bias [+ addRow[addIdx]]),
// stored directly or atomically scattered by row (scatterC).
// grid = (ceil(M/64), Ncols/128)
// ---------------------------------------------------------------------------
template <bool FAST>
__global__ void __launch_bounds__(256) gemm_wmma_k(
    const float* __restrict__ A0, const int* __restrict__ gatherA, int A0cols,
    const float* __restrict__ A1, int A1cols,
    const float* __restrict__ B, int ldb,
    const float* __restrict__ bias,
    const float* __restrict__ addRow, const int* __restrict__ addIdx, int addLd,
    int act,
    float* __restrict__ C, int ldc, const int* __restrict__ scatterC,
    int M)
{
    const int rowbase = blockIdx.x * (16 * MT);
    const int cb = blockIdx.y * 128;
    v8f acc[MT];
    wmma_rowtile<FAST>(A0, gatherA, A0cols, A1, A1cols, B, ldb, cb, rowbase, M, acc);

    const int lane = threadIdx.x & 31, wave = threadIdx.x >> 5;
    const int nIdx = lane & 15, half = lane >> 4;
    const int col = cb + wave * 16 + nIdx;
    const float bv = bias ? bias[col] : 0.f;
#pragma unroll
    for (int mt = 0; mt < MT; ++mt)
#pragma unroll
    for (int r = 0; r < 8; ++r) {
        int row = rowbase + mt * 16 + r + 8 * half;
        if (row < M) {
            float x = acc[mt][r] + bv;
            if (addRow) {
                int ai = addIdx ? addIdx[row] : row;
                x += addRow[(size_t)ai * addLd + col];
            }
            x = activate(x, act);
            if (scatterC) atomicAdd(&C[(size_t)scatterC[row] * ldc + col], x);
            else          C[(size_t)row * ldc + col] = x;
        }
    }
}

// ---------------------------------------------------------------------------
// Attention logits: a[k,n] = sum_h tanh(v@Wam_k + b)[n,h] * a_super[k,batch[n],h]
//                          * Wbmm_w[k,h]  + Wbmm_b[k]
// grid = (ceil(N/64), K)
// ---------------------------------------------------------------------------
__global__ void __launch_bounds__(256) attn_logit_k(
    const float* __restrict__ v,
    const float* __restrict__ Wam_w, const float* __restrict__ Wam_b,
    const float* __restrict__ Wbmm_w, const float* __restrict__ Wbmm_b,
    const float* __restrict__ a_super, const int* __restrict__ batch,
    float* __restrict__ logits)
{
    const int rowbase = blockIdx.x * (16 * MT);
    const int kh = blockIdx.y;
    __shared__ float red[8][16 * MT];

    v8f acc[MT];
    wmma_rowtile<true>(v, nullptr, Hh, nullptr, 0,
                       Wam_w + (size_t)kh * Hh * Hh, Hh, 0, rowbase, Nv, acc);

    const int tid = threadIdx.x;
    const int lane = tid & 31, wave = tid >> 5;
    const int nIdx = lane & 15, half = lane >> 4;
    const int col = wave * 16 + nIdx;
    const float bv = Wam_b[kh * Hh + col];
    const float wb = Wbmm_w[kh * Hh + col];
#pragma unroll
    for (int mt = 0; mt < MT; ++mt)
#pragma unroll
    for (int r = 0; r < 8; ++r) {
        int row = rowbase + mt * 16 + r + 8 * half;
        int rc = row < Nv ? row : Nv - 1;
        int g = batch[rc];
        float x = tanhf(acc[mt][r] + bv);
        float contrib = (row < Nv)
            ? x * a_super[((size_t)kh * Gg + g) * Hh + col] * wb : 0.f;
        contrib += __shfl_xor(contrib, 1, 32);
        contrib += __shfl_xor(contrib, 2, 32);
        contrib += __shfl_xor(contrib, 4, 32);
        contrib += __shfl_xor(contrib, 8, 32);
        if (nIdx == 0) red[wave][mt * 16 + r + 8 * half] = contrib;
    }
    __syncthreads();
    if (tid < 16 * MT) {
        float t = Wbmm_b[kh];
#pragma unroll
        for (int w2 = 0; w2 < 8; ++w2) t += red[w2][tid];
        int row = rowbase + tid;
        if (row < Nv) logits[(size_t)kh * Nv + row] = t;
    }
}

// ---------------------------------------------------------------------------
// Attention-weighted scatter: m2s_in[batch[n], k*H + h] += attn[k,n]*(v@Wm_k + b)
// grid = (ceil(N/64), K)
// ---------------------------------------------------------------------------
__global__ void __launch_bounds__(256) attn_scatter_k(
    const float* __restrict__ v,
    const float* __restrict__ Wm_w, const float* __restrict__ Wm_b,
    const float* __restrict__ logits, const float* __restrict__ gmax,
    const float* __restrict__ gsum, const int* __restrict__ batch,
    float* __restrict__ m2s_in)
{
    const int rowbase = blockIdx.x * (16 * MT);
    const int kh = blockIdx.y;
    v8f acc[MT];
    wmma_rowtile<true>(v, nullptr, Hh, nullptr, 0,
                       Wm_w + (size_t)kh * Hh * Hh, Hh, 0, rowbase, Nv, acc);
    const int lane = threadIdx.x & 31, wave = threadIdx.x >> 5;
    const int nIdx = lane & 15, half = lane >> 4;
    const int col = wave * 16 + nIdx;
    const float bv = Wm_b[kh * Hh + col];
#pragma unroll
    for (int mt = 0; mt < MT; ++mt)
#pragma unroll
    for (int r = 0; r < 8; ++r) {
        int row = rowbase + mt * 16 + r + 8 * half;
        if (row < Nv) {
            int g = batch[row];
            float attn = __expf(logits[(size_t)kh * Nv + row] - gmax[kh * Gg + g])
                       / (gsum[kh * Gg + g] + 1e-6f);
            atomicAdd(&m2s_in[(size_t)g * (Kh * Hh) + kh * Hh + col],
                      attn * (acc[mt][r] + bv));
        }
    }
}

// ---------------------------------------------------------------------------
// Segment softmax stats (atomic max / exp-sum over batch segments)
// ---------------------------------------------------------------------------
__global__ void seg_max_k(const float* __restrict__ logits,
                          const int* __restrict__ batch, float* __restrict__ gmax)
{
    int i = blockIdx.x * blockDim.x + threadIdx.x;
    if (i >= Kh * Nv) return;
    int kh = i / Nv, n = i - kh * Nv;
    float val = logits[i];
    float* addr = &gmax[kh * Gg + batch[n]];
    if (val >= 0.f) atomicMax((int*)addr, __float_as_int(val));
    else            atomicMin((unsigned int*)addr, __float_as_uint(val));
}

__global__ void seg_expsum_k(const float* __restrict__ logits,
                             const int* __restrict__ batch,
                             const float* __restrict__ gmax,
                             float* __restrict__ gsum)
{
    int i = blockIdx.x * blockDim.x + threadIdx.x;
    if (i >= Kh * Nv) return;
    int kh = i / Nv, n = i - kh * Nv;
    int g = batch[n];
    atomicAdd(&gsum[kh * Gg + g], __expf(logits[i] - gmax[kh * Gg + g]));
}

// ---------------------------------------------------------------------------
// Elementwise kernels
// ---------------------------------------------------------------------------
__global__ void fill_k(float* __restrict__ p, float v, int n)
{
    int i = blockIdx.x * blockDim.x + threadIdx.x;
    if (i < n) p[i] = v;
}

// out = (1 - z) * a + z * b[gidx ? gidx[row] : row]
__global__ void fuse_gate_k(const float* __restrict__ z, const float* __restrict__ a,
                            const float* __restrict__ b, const int* __restrict__ gidx,
                            float* __restrict__ out, int M)
{
    int i = blockIdx.x * blockDim.x + threadIdx.x;
    if (i >= M * Hh) return;
    int row = i >> 7, c = i & 127;
    int br = gidx ? gidx[row] : row;
    float zz = z[i];
    out[i] = (1.f - zz) * a[i] + zz * b[(size_t)br * Hh + c];
}

// GRU cell combine: gi,gh are [M,3H]; out = (1-z)*n + z*h_prev
__global__ void gru_k(const float* __restrict__ gi, const float* __restrict__ gh,
                      const float* __restrict__ hp, float* __restrict__ out, int M)
{
    int i = blockIdx.x * blockDim.x + threadIdx.x;
    if (i >= M * Hh) return;
    int row = i >> 7, c = i & 127;
    size_t o = (size_t)row * H3 + c;
    float r = sigmf(gi[o]          + gh[o]);
    float z = sigmf(gi[o + Hh]     + gh[o + Hh]);
    float n = tanhf(gi[o + 2 * Hh] + r * gh[o + 2 * Hh]);
    out[i] = (1.f - z) * n + z * hp[i];
}

// ---------------------------------------------------------------------------
extern "C" void kernel_launch(void* const* d_in, const int* in_sizes, int n_in,
                              void* d_out, int out_size, void* d_ws, size_t ws_size,
                              hipStream_t stream)
{
    (void)in_sizes; (void)n_in; (void)out_size; (void)ws_size;
    const float* v      = (const float*)d_in[0];
    const float* s      = (const float*)d_in[1];
    const float* ea     = (const float*)d_in[2];
    const int*   ei     = (const int*)d_in[3];
    const int*   batch  = (const int*)d_in[4];
    const float* Wam_w  = (const float*)d_in[5];  const float* Wam_b = (const float*)d_in[6];
    const float* Was_w  = (const float*)d_in[7];  const float* Was_b = (const float*)d_in[8];
    const float* Wm_w   = (const float*)d_in[9];  const float* Wm_b  = (const float*)d_in[10];
    const float* Wbmm_w = (const float*)d_in[11]; const float* Wbmm_b= (const float*)d_in[12];
    const float* Wsup_w = (const float*)d_in[13]; const float* Wsup_b= (const float*)d_in[14];
    const float* Wm2s_w = (const float*)d_in[15]; const float* Wm2s_b= (const float*)d_in[16];
    const float* Ws2m_w = (const float*)d_in[17]; const float* Ws2m_b= (const float*)d_in[18];
    const float* Wzm1_w = (const float*)d_in[19]; const float* Wzm1_b= (const float*)d_in[20];
    const float* Wzm2_w = (const float*)d_in[21]; const float* Wzm2_b= (const float*)d_in[22];
    const float* Wzs1_w = (const float*)d_in[23]; const float* Wzs1_b= (const float*)d_in[24];
    const float* Wzs2_w = (const float*)d_in[25]; const float* Wzs2_b= (const float*)d_in[26];
    const float* U2_w   = (const float*)d_in[27]; const float* U2_b  = (const float*)d_in[28];
    const float* U1_w   = (const float*)d_in[29]; const float* U1_b  = (const float*)d_in[30];
    const float* gm_wih = (const float*)d_in[31]; const float* gm_bih= (const float*)d_in[32];
    const float* gm_whh = (const float*)d_in[33]; const float* gm_bhh= (const float*)d_in[34];
    const float* gs_wih = (const float*)d_in[35]; const float* gs_bih= (const float*)d_in[36];
    const float* gs_whh = (const float*)d_in[37]; const float* gs_bhh= (const float*)d_in[38];
    const int* src = ei;
    const int* dst = ei + Ee;

    // ---- scratch layout ----
    float* w = (float*)d_ws;
    float* a_super = w; w += (size_t)Kh * Gg * Hh;
    float* s2m     = w; w += Gg * Hh;
    float* ssup    = w; w += Gg * Hh;
    float* zm2s    = w; w += Gg * Hh;
    float* m2s     = w; w += Gg * Hh;
    float* t1      = w; w += Gg * Hh;
    float* zsup    = w; w += Gg * Hh;
    float* hsup    = w; w += Gg * Hh;
    float* giS     = w; w += Gg * H3;
    float* ghS     = w; w += Gg * H3;
    float* logits  = w; w += (size_t)Kh * Nv;
    float* gmax    = w; w += Kh * Gg;
    float* gsum    = w; w += Kh * Gg;
    float* m2s_in  = w; w += (size_t)Gg * Kh * Hh;
    float* agg     = w; w += (size_t)Nv * Hh;
    float* mself   = w; w += (size_t)Nv * Hh;
    float* zmain   = w; w += (size_t)Nv * Hh;
    float* hmain   = w; w += (size_t)Nv * Hh;
    float* giM     = w; w += (size_t)Nv * H3;
    float* ghM     = w; w += (size_t)Nv * H3;

    float* vout = (float*)d_out;
    float* sout = vout + (size_t)Nv * Hh;

    const dim3 blk(256);
    const int gN = (Nv + 63) / 64;         // 782
    const int gG = (Gg + 63) / 64;         // 16
    const int gE = Ee / 64;                // 10000

    // ---- init accumulation buffers ----
    fill_k<<<(Nv * Hh + 255) / 256, blk, 0, stream>>>(agg, 0.f, Nv * Hh);
    fill_k<<<(Gg * Kh * Hh + 255) / 256, blk, 0, stream>>>(m2s_in, 0.f, Gg * Kh * Hh);
    fill_k<<<(Kh * Gg + 255) / 256, blk, 0, stream>>>(gmax, -3.0e38f, Kh * Gg);
    fill_k<<<(Kh * Gg + 255) / 256, blk, 0, stream>>>(gsum, 0.f, Kh * Gg);

    // ---- small super-node GEMMs (branchless staging) ----
    for (int k = 0; k < Kh; ++k)
        gemm_wmma_k<true><<<dim3(gG, 1), blk, 0, stream>>>(
            s, nullptr, Hh, nullptr, 0, Was_w + (size_t)k * Hh * Hh, Hh,
            Was_b + k * Hh, nullptr, nullptr, 0, 1,
            a_super + (size_t)k * Gg * Hh, Hh, nullptr, Gg);
    gemm_wmma_k<true><<<dim3(gG, 1), blk, 0, stream>>>(s, nullptr, Hh, nullptr, 0, Ws2m_w, Hh,
        Ws2m_b, nullptr, nullptr, 0, 1, s2m, Hh, nullptr, Gg);
    gemm_wmma_k<true><<<dim3(gG, 1), blk, 0, stream>>>(s, nullptr, Hh, nullptr, 0, Wsup_w, Hh,
        Wsup_b, nullptr, nullptr, 0, 1, ssup, Hh, nullptr, Gg);
    gemm_wmma_k<true><<<dim3(gG, 1), blk, 0, stream>>>(s2m, nullptr, Hh, nullptr, 0, Wzm2_w, Hh,
        Wzm2_b, nullptr, nullptr, 0, 0, zm2s, Hh, nullptr, Gg);

    // ---- attention: logits -> softmax stats -> weighted scatter ----
    attn_logit_k<<<dim3(gN, Kh), blk, 0, stream>>>(v, Wam_w, Wam_b, Wbmm_w, Wbmm_b,
                                                   a_super, batch, logits);
    seg_max_k<<<(Kh * Nv + 255) / 256, blk, 0, stream>>>(logits, batch, gmax);
    seg_expsum_k<<<(Kh * Nv + 255) / 256, blk, 0, stream>>>(logits, batch, gmax, gsum);
    attn_scatter_k<<<dim3(gN, Kh), blk, 0, stream>>>(v, Wm_w, Wm_b, logits, gmax, gsum,
                                                     batch, m2s_in);
    // main_to_super = tanh([G, K*H] @ Wm2s)
    gemm_wmma_k<true><<<dim3(gG, 1), blk, 0, stream>>>(m2s_in, nullptr, Kh * Hh, nullptr, 0,
        Wm2s_w, Hh, Wm2s_b, nullptr, nullptr, 0, 1, m2s, Hh, nullptr, Gg);

    // ---- WLN edge messages: gather-GEMM + LeakyReLU + scatter-add ----
    gemm_wmma_k<false><<<dim3(gE, 1), blk, 0, stream>>>(
        v, src, Hh, ea, BFe, U2_w, Hh, U2_b, nullptr, nullptr, 0, 3,
        agg, Hh, dst, Ee);

    // ---- vertex path ----
    gemm_wmma_k<false><<<dim3(gN, 1), blk, 0, stream>>>(v, nullptr, Hh, agg, Hh, U1_w, Hh,
        U1_b, nullptr, nullptr, 0, 0, mself, Hh, nullptr, Nv);
    gemm_wmma_k<true><<<dim3(gN, 1), blk, 0, stream>>>(mself, nullptr, Hh, nullptr, 0, Wzm1_w, Hh,
        Wzm1_b, zm2s, batch, Hh, 2, zmain, Hh, nullptr, Nv);
    fuse_gate_k<<<(Nv * Hh + 255) / 256, blk, 0, stream>>>(zmain, mself, s2m, batch, hmain, Nv);
    gemm_wmma_k<true><<<dim3(gN, 3), blk, 0, stream>>>(hmain, nullptr, Hh, nullptr, 0, gm_wih, H3,
        gm_bih, nullptr, nullptr, 0, 0, giM, H3, nullptr, Nv);
    gemm_wmma_k<true><<<dim3(gN, 3), blk, 0, stream>>>(v, nullptr, Hh, nullptr, 0, gm_whh, H3,
        gm_bhh, nullptr, nullptr, 0, 0, ghM, H3, nullptr, Nv);
    gru_k<<<(Nv * Hh + 255) / 256, blk, 0, stream>>>(giM, ghM, v, vout, Nv);

    // ---- super path ----
    gemm_wmma_k<true><<<dim3(gG, 1), blk, 0, stream>>>(ssup, nullptr, Hh, nullptr, 0, Wzs1_w, Hh,
        Wzs1_b, nullptr, nullptr, 0, 0, t1, Hh, nullptr, Gg);
    gemm_wmma_k<true><<<dim3(gG, 1), blk, 0, stream>>>(m2s, nullptr, Hh, nullptr, 0, Wzs2_w, Hh,
        Wzs2_b, t1, nullptr, Hh, 2, zsup, Hh, nullptr, Gg);
    fuse_gate_k<<<(Gg * Hh + 255) / 256, blk, 0, stream>>>(zsup, ssup, m2s, nullptr, hsup, Gg);
    gemm_wmma_k<true><<<dim3(gG, 3), blk, 0, stream>>>(hsup, nullptr, Hh, nullptr, 0, gs_wih, H3,
        gs_bih, nullptr, nullptr, 0, 0, giS, H3, nullptr, Gg);
    gemm_wmma_k<true><<<dim3(gG, 3), blk, 0, stream>>>(s, nullptr, Hh, nullptr, 0, gs_whh, H3,
        gs_bhh, nullptr, nullptr, 0, 0, ghS, H3, nullptr, Gg);
    gru_k<<<(Gg * Hh + 255) / 256, blk, 0, stream>>>(giS, ghS, s, sout, Gg);
}
